// Attention_61856118997190
// MI455X (gfx1250) — matrix operand order repaired and checked by reading, not compiled, not run on hardware
//
#include <hip/hip_runtime.h>

typedef __attribute__((ext_vector_type(16))) _Float16 v16h;
typedef __attribute__((ext_vector_type(8)))  _Float16 h8;
typedef __attribute__((ext_vector_type(4)))  _Float16 h4;
typedef __attribute__((ext_vector_type(8)))  float    v8f;

__device__ __forceinline__ v8f wmma_f16(v16h a, v16h b, v8f c) {
    // D = A(16x32 f16) * B(32x16 f16) + C(16x16 f32)
    return __builtin_amdgcn_wmma_f32_16x16x32_f16(false, a, false, b,
                                                  (short)0, c, false, false);
}

__device__ __forceinline__ v16h frag16(h8 lo, h8 hi) {
    return __builtin_shufflevector(lo, hi, 0, 1, 2, 3, 4, 5, 6, 7,
                                           8, 9, 10, 11, 12, 13, 14, 15);
}

// max-reduction across the 16-lane half-groups (xor masks < 16 stay in-half)
__device__ __forceinline__ float redmax16(float v) {
    #pragma unroll
    for (int m = 8; m >= 1; m >>= 1) v = fmaxf(v, __shfl_xor(v, m, 32));
    return v;
}

__device__ __forceinline__ h4 ld4h(const float* p) {
    float4 v = *(const float4*)p;
    h4 r;
    r[0] = (_Float16)v.x; r[1] = (_Float16)v.y;
    r[2] = (_Float16)v.z; r[3] = (_Float16)v.w;
    return r;
}

// ---------------------------------------------------------------------------
// Tiled WMMA GEMM: C[M,N] = A[M,K] * B[K,N] (+bias).
// Block tile 128x64, 4 waves, K-step 32; each wave register-blocks 2(M)x4(N)
// fragments -> 8 WMMAs per K-step over one A/B stage (high math : staging
// ratio). fp16 A tiles are staged with GLOBAL_LOAD_ASYNC_TO_LDS_B128 (raw
// copy, per-lane global + LDS addresses); fp32 A tiles convert through VGPRs.
// B staged transposed so B-fragments are two contiguous 128-bit LDS reads.
// ---------------------------------------------------------------------------
template <typename TA, typename TC, bool BIAS>
__global__ __launch_bounds__(128)
void gemm_wmma(const TA* __restrict__ A, const float* __restrict__ B,
               const float* __restrict__ bias, TC* __restrict__ C,
               int M, int N, int K)
{
    __shared__ __align__(16) _Float16 As[128][40];  // [m][k], 80B pitch
    __shared__ __align__(16) _Float16 Bt[64][40];   // [n][k], 80B pitch

    const int tid  = threadIdx.x;
    const int lane = tid & 31;
    const int wave = tid >> 5;
    const int l15  = lane & 15;
    const int lh   = lane >> 4;       // 0/1 half-group
    const int n0   = blockIdx.x * 64;
    const int m0   = blockIdx.y * 128;
    const int mW   = 32 * wave;       // wave's 32-row strip in the tile

    v8f acc[2][4];
    #pragma unroll
    for (int s = 0; s < 2; ++s)
        #pragma unroll
        for (int t = 0; t < 4; ++t)
            #pragma unroll
            for (int r = 0; r < 8; ++r) acc[s][t][r] = 0.0f;

    for (int k0 = 0; k0 < K; k0 += 32) {
        // ---- stage A tile 128x32 ----
        if constexpr (sizeof(TA) == 2) {
            // fp16: raw async DMA global -> LDS, 512 x 16B, 4 per thread
            #pragma unroll
            for (int it = 0; it < 4; ++it) {
                int e = tid + 128 * it;          // 0..511
                int r = e >> 2, c = (e & 3) * 8; // 4 x 16B per 32-half row
                unsigned long long src =
                    (unsigned long long)(const void*)&A[(size_t)(m0 + r) * K + (k0 + c)];
                unsigned dst = (unsigned)(unsigned long long)(const void*)&As[r][c];
                asm volatile("global_load_async_to_lds_b128 %0, %1, off"
                             :: "v"(dst), "v"(src) : "memory");
            }
        } else {
            // fp32: 128-bit loads, convert, 64-bit LDS stores
            #pragma unroll
            for (int it = 0; it < 8; ++it) {
                int e = tid + 128 * it;          // 0..1023
                int r = e >> 3, c = (e & 7) * 4;
                *(h4*)&As[r][c] = ld4h(&A[(size_t)(m0 + r) * K + (k0 + c)]);
            }
        }
        // ---- stage B tile 32x64 transposed ----
        #pragma unroll
        for (int it = 0; it < 4; ++it) {
            int e = tid + 128 * it;              // 0..511
            int r = e >> 4, c = (e & 15) * 4;    // r = k, c = n
            float4 v = *(const float4*)&B[(size_t)(k0 + r) * N + (n0 + c)];
            Bt[c + 0][r] = (_Float16)v.x;
            Bt[c + 1][r] = (_Float16)v.y;
            Bt[c + 2][r] = (_Float16)v.z;
            Bt[c + 3][r] = (_Float16)v.w;
        }
        if constexpr (sizeof(TA) == 2)
            asm volatile("s_wait_asynccnt 0x0" ::: "memory");
        __syncthreads();

        // A fragments (16x32 f16, ISA 7.12.2): M = lane%16,
        // elements 0..7 -> K = 8*lh + 0..7, elements 8..15 -> K = 8*lh + 16..23
        v16h afr[2];
        #pragma unroll
        for (int s = 0; s < 2; ++s) {
            const _Float16* arow = &As[mW + 16 * s + l15][0];
            afr[s] = frag16(*(const h8*)(arow + 8 * lh),
                            *(const h8*)(arow + 8 * lh + 16));
        }

        #pragma unroll
        for (int t = 0; t < 4; ++t) {
            // B fragment: N = lane%16, K(e) = 8*lh + e (e<8), 8*lh + 16 + (e-8) (e>=8)
            const _Float16* brow = &Bt[16 * t + l15][8 * lh];
            v16h bfr = frag16(*(const h8*)brow, *(const h8*)(brow + 16));
            acc[0][t] = wmma_f16(afr[0], bfr, acc[0][t]);
            acc[1][t] = wmma_f16(afr[1], bfr, acc[1][t]);
        }
        __syncthreads();
    }

    // C/D layout: col = lane%16, row = r + 8*lh
    #pragma unroll
    for (int s = 0; s < 2; ++s)
        #pragma unroll
        for (int t = 0; t < 4; ++t)
            #pragma unroll
            for (int r = 0; r < 8; ++r) {
                int row = m0 + mW + 16 * s + r + 8 * lh;
                int col = n0 + 16 * t + l15;
                float v = acc[s][t][r];
                if (BIAS) v += bias[col];
                C[(size_t)row * N + col] = (TC)v;
            }
}

// ---------------------------------------------------------------------------
// Flash attention: one wave per (b, h, 16-row Q tile). 32-key steps:
//   S(16x32) = (Q*scale)(16x64) Kt   -> 2 subtiles x 2 chained K=32 WMMAs
//   online softmax; row max via ds_bpermute reduction, row SUM via an extra
//   WMMA against an all-ones B matrix (every C column = row sum)
//   O = O*alpha + P(16x32) V(32x64)  -> 4 subtiles x 1 K=32 WMMA
// V tile staged to LDS via GLOBAL_LOAD_ASYNC_TO_LDS_B128 (ASYNCcnt path);
// P relaid C-layout -> A-layout through 1KB LDS.
// ---------------------------------------------------------------------------
#define NSEQ   1024
#define NHEAD  12
#define DHEAD  64
#define QKVLD  2304
#define INNERD 768

__global__ __launch_bounds__(32)
void attn_wmma(const _Float16* __restrict__ qkv, _Float16* __restrict__ outh)
{
    __shared__ __align__(16) _Float16 Pl[16][40];   // [qrow][key], 80B pitch
    __shared__ __align__(16) _Float16 Vl[32][72];   // [key][d], 144B pitch

    const int lane = threadIdx.x;
    const int l15  = lane & 15;
    const int lh   = lane >> 4;
    const int q0   = blockIdx.x * 16;
    const int h    = blockIdx.y;
    const int b    = blockIdx.z;

    const size_t rowbase = (size_t)b * NSEQ;
    const int qcol = h * DHEAD;
    const int kcol = INNERD + h * DHEAD;
    const int vcol = 2 * INNERD + h * DHEAD;

    // all-ones B fragment: P x ones -> row sums of P in every C column
    v16h ones;
    #pragma unroll
    for (int e = 0; e < 16; ++e) ones[e] = (_Float16)1.0f;

    // Q A-fragments (d-halves s=0: 0..31, s=1: 32..63); fold in softmax scale
    v16h qa[2];
    #pragma unroll
    for (int s = 0; s < 2; ++s) {
        const _Float16* qrow = qkv + (rowbase + q0 + l15) * QKVLD + qcol + 32 * s;
        v16h q = frag16(*(const h8*)(qrow + 8 * lh),
                        *(const h8*)(qrow + 8 * lh + 16));
        qa[s] = q * (_Float16)0.125f;   // 1/sqrt(64)
    }

    v8f o[4];
    float mrow[8], lrow[8];
    #pragma unroll
    for (int t = 0; t < 4; ++t)
        #pragma unroll
        for (int r = 0; r < 8; ++r) o[t][r] = 0.0f;
    #pragma unroll
    for (int r = 0; r < 8; ++r) { mrow[r] = -1e30f; lrow[r] = 0.0f; }

    for (int j0 = 0; j0 < NSEQ; j0 += 32) {
        // async-stage V tile (32 keys x 64 d): one row per lane, 8 x 16B.
        // INST_OFFSET adds to BOTH the global and LDS address (ISA 08 §4.4),
        // so one base pair + 8 literal offsets copies the whole row.
        {
            unsigned long long src =
                (unsigned long long)(const void*)(qkv + (rowbase + j0 + lane) * QKVLD + vcol);
            unsigned vdst = (unsigned)(unsigned long long)(const void*)&Vl[lane][0];
            asm volatile(
                "global_load_async_to_lds_b128 %0, %1, off\n\t"
                "global_load_async_to_lds_b128 %0, %1, off offset:16\n\t"
                "global_load_async_to_lds_b128 %0, %1, off offset:32\n\t"
                "global_load_async_to_lds_b128 %0, %1, off offset:48\n\t"
                "global_load_async_to_lds_b128 %0, %1, off offset:64\n\t"
                "global_load_async_to_lds_b128 %0, %1, off offset:80\n\t"
                "global_load_async_to_lds_b128 %0, %1, off offset:96\n\t"
                "global_load_async_to_lds_b128 %0, %1, off offset:112"
                :: "v"(vdst), "v"(src) : "memory");
        }

        // S = (Q*scale) * K^T for the 32-key block (2 column subtiles)
        v8f sfr[2];
        #pragma unroll
        for (int t = 0; t < 2; ++t) {
            const _Float16* krow =
                qkv + (rowbase + j0 + 16 * t + l15) * QKVLD + kcol + 8 * lh;
            v16h kb0 = frag16(*(const h8*)krow,        *(const h8*)(krow + 16));
            v16h kb1 = frag16(*(const h8*)(krow + 32), *(const h8*)(krow + 48));
            v8f a;
            #pragma unroll
            for (int r = 0; r < 8; ++r) a[r] = 0.0f;
            a = wmma_f16(qa[0], kb0, a);
            a = wmma_f16(qa[1], kb1, a);
            sfr[t] = a;
        }

        // online softmax: row max via shuffle reduce; exp in fp32
        float alpha[8], p0a[8], p1a[8];
        #pragma unroll
        for (int r = 0; r < 8; ++r) {
            float s0 = sfr[0][r];
            float s1 = sfr[1][r];
            float mx   = redmax16(fmaxf(s0, s1));
            float mnew = fmaxf(mrow[r], mx);
            alpha[r] = __expf(mrow[r] - mnew);
            mrow[r]  = mnew;
            p0a[r] = __expf(s0 - mnew);
            p1a[r] = __expf(s1 - mnew);
        }

        // P: C-fragment layout -> LDS -> A-fragment layout
        __syncthreads();
        #pragma unroll
        for (int r = 0; r < 8; ++r) {
            Pl[r + 8 * lh][l15]      = (_Float16)p0a[r];
            Pl[r + 8 * lh][16 + l15] = (_Float16)p1a[r];
        }
        __syncthreads();

        const _Float16* prow = &Pl[l15][0];
        v16h pa = frag16(*(const h8*)(prow + 8 * lh),
                         *(const h8*)(prow + 8 * lh + 16));

        // row sums of P via WMMA against all-ones B
        v8f rs;
        #pragma unroll
        for (int r = 0; r < 8; ++r) rs[r] = 0.0f;
        rs = wmma_f16(pa, ones, rs);

        // V tile must be resident before B-fragment reads
        asm volatile("s_wait_asynccnt 0x0" ::: "memory");

        // O = O*alpha + P*V  (4 output subtiles of 16 along d)
        #pragma unroll
        for (int t = 0; t < 4; ++t) {
            #pragma unroll
            for (int r = 0; r < 8; ++r) o[t][r] *= alpha[r];
            v16h vb;
            #pragma unroll
            for (int e = 0; e < 16; ++e)
                vb[e] = Vl[(e & 7) + 8 * lh + 2 * (e & 8)][16 * t + l15];
            o[t] = wmma_f16(pa, vb, o[t]);
        }

        #pragma unroll
        for (int r = 0; r < 8; ++r) lrow[r] = lrow[r] * alpha[r] + rs[r];

        __syncthreads();   // protect Vl/Pl before next iteration's writes
    }

    // normalize and write out in [B*N, H*D] layout (fp16 for next GEMM)
    #pragma unroll
    for (int r = 0; r < 8; ++r) {
        float inv = 1.0f / lrow[r];
        int row = q0 + r + 8 * lh;
        #pragma unroll
        for (int t = 0; t < 4; ++t) {
            float v = o[t][r] * inv;
            outh[(rowbase + row) * INNERD + h * DHEAD + 16 * t + l15] = (_Float16)v;
        }
    }
}

// ---------------------------------------------------------------------------
extern "C" void kernel_launch(void* const* d_in, const int* in_sizes, int n_in,
                              void* d_out, int out_size, void* d_ws, size_t ws_size,
                              hipStream_t stream) {
    (void)in_sizes; (void)n_in; (void)out_size; (void)ws_size;

    const float* x     = (const float*)d_in[0];   // [4,1024,768]
    const float* w_qkv = (const float*)d_in[1];   // [768,2304]
    const float* w_out = (const float*)d_in[2];   // [768,768]
    const float* b_out = (const float*)d_in[3];   // [768]
    float* out = (float*)d_out;                   // [4,1024,768]

    const int M = 4 * NSEQ;                       // 4096 rows
    _Float16* qkv_h  = (_Float16*)d_ws;                       // 4096*2304 f16
    _Float16* attn_h = qkv_h + (size_t)M * QKVLD;             // 4096*768  f16

    // 1) QKV projection: fp32 in, fp16 intermediate out (128x64 tiles)
    gemm_wmma<float, _Float16, false>
        <<<dim3(QKVLD / 64, M / 128), dim3(128), 0, stream>>>(
            x, w_qkv, nullptr, qkv_h, M, QKVLD, INNERD);

    // 2) Flash attention over all (b, h, q-tile)
    attn_wmma<<<dim3(NSEQ / 16, NHEAD, 4), dim3(32), 0, stream>>>(qkv_h, attn_h);

    // 3) Output projection + bias: fp16 in (async-staged), fp32 out
    gemm_wmma<_Float16, float, true>
        <<<dim3(INNERD / 64, M / 128), dim3(128), 0, stream>>>(
            attn_h, w_out, b_out, out, M, INNERD, INNERD);
}
